// DeterministicDecoder_65730179498244
// MI455X (gfx1250) — matrix operand order repaired and checked by reading, not compiled, not run on hardware
//
#include <hip/hip_runtime.h>
#include <hip/hip_bf16.h>

typedef __attribute__((ext_vector_type(16))) _Float16 v16h;
typedef __attribute__((ext_vector_type(8)))  float    v8f;

#define BB   16384
#define FEAT 26
#define VOC  100000
#define ED   16
#define REPD 64
#define DDEN 13
#define KPAD 512      // 493 padded to 512
#define H1D  256
#define H2D  128
#define DIN  493
#define KB1  (KPAD / 32)   // 16 k-tiles, GEMM1
#define NT1  (H1D / 16)    // 16 n-tiles, GEMM1
#define KB2  (H1D / 32)    // 8 k-tiles, GEMM2
#define NT2  (H2D / 16)    // 8 n-tiles, GEMM2

__device__ __forceinline__ float wave_sum32(float v) {
    v += __shfl_xor(v, 1, 32);
    v += __shfl_xor(v, 2, 32);
    v += __shfl_xor(v, 4, 32);
    v += __shfl_xor(v, 8, 32);
    v += __shfl_xor(v, 16, 32);
    return v;
}

// ---------------------------------------------------------------------------
// Kernel 1: per-sample gather + FM(1st,2nd order) + pack dnn_in as f16 [B,512]
// One wave per sample; 8 waves / block.
// dnn_in: [0..415]=emb(26x16), [416..428]=dense_x, [429..492]=rep, [493..511]=0
// ---------------------------------------------------------------------------
__global__ void gather_pack_fm(const float* __restrict__ rep,
                               const float* __restrict__ dx,
                               const int*   __restrict__ ids,
                               const float* __restrict__ fm1t,
                               const float* __restrict__ fm2t,
                               const float* __restrict__ fm1w,
                               const float* __restrict__ fm1b,
                               _Float16*    __restrict__ dnn_in,
                               float*       __restrict__ fm_part) {
    const int lane = threadIdx.x & 31;
    const int b    = blockIdx.x * 8 + (threadIdx.x >> 5);
    const int h    = lane >> 4;
    const int d    = lane & 15;
    const int* bid = ids + (size_t)b * FEAT;
    _Float16* drow = dnn_in + (size_t)b * KPAD;

    float s = 0.f, sq = 0.f;
    const int f0 = h * 13;
#pragma unroll
    for (int j = 0; j < 13; ++j) {
        const int f  = f0 + j;
        const int id = bid[f];
        const float v = fm2t[((size_t)f * VOC + (size_t)id) * ED + d];
        drow[f * ED + d] = (_Float16)v;
        s  += v;
        sq += v * v;
    }
    s  += __shfl_xor(s, 16, 32);
    sq += __shfl_xor(sq, 16, 32);
    const float fm2v = (lane < 16) ? (s * s - sq) : 0.f;

    const int fl = (lane < FEAT) ? lane : 0;
    float f1 = fm1t[(size_t)fl * VOC + (size_t)bid[fl]];
    if (lane >= FEAT) f1 = 0.f;

    float dot = 0.f;
#pragma unroll
    for (int t = 0; t < 3; ++t) {
        const int jj = lane + t * 32;            // 0..95 covers 416..511
        float v = 0.f;
        if (jj < DDEN)             v = dx[(size_t)b * DDEN + jj];
        else if (jj < DDEN + REPD) v = rep[(size_t)b * REPD + (jj - DDEN)];
        drow[FEAT * ED + jj] = (_Float16)v;
        if (jj < DDEN + REPD) dot += v * fm1w[jj];
    }

    const float tot = wave_sum32(0.5f * fm2v + f1 + dot);
    if (lane == 0) fm_part[b] = tot + fm1b[0];
}

// ---------------------------------------------------------------------------
// Weight repack f32 -> f16 in WMMA B-fragment order:
//   Bp[((kb*NT + nt)*32 + lane)*16 + e] = W[kb*32 + (e>>3)*16 + (lane>>4)*8 + (e&7)][nt*16 + (lane&15)]
// so each lane's 16-half fragment is one contiguous, 32B-aligned load.
// ---------------------------------------------------------------------------
__global__ void pack_w1_frag(const float* __restrict__ w, _Float16* __restrict__ wp) {
    const int o = blockIdx.x * 256 + threadIdx.x;        // over 16*16*32*16
    const int e    = o & 15;
    const int lane = (o >> 4) & 31;
    const int nt   = (o >> 9) & (NT1 - 1);
    const int kb   = o >> 13;
    const int k = kb * 32 + ((e >> 3) << 4) + (lane >> 4) * 8 + (e & 7);
    const int n = nt * 16 + (lane & 15);
    wp[o] = (k < DIN) ? (_Float16)w[k * H1D + n] : (_Float16)0.f;
}
__global__ void pack_w2_frag(const float* __restrict__ w, _Float16* __restrict__ wp) {
    const int o = blockIdx.x * 256 + threadIdx.x;        // over 8*8*32*16
    const int e    = o & 15;
    const int lane = (o >> 4) & 31;
    const int nt   = (o >> 9) & (NT2 - 1);
    const int kb   = o >> 12;
    const int k = kb * 32 + ((e >> 3) << 4) + (lane >> 4) * 8 + (e & 7);
    const int n = nt * 16 + (lane & 15);
    wp[o] = (_Float16)w[k * H2D + n];
}

// ---------------------------------------------------------------------------
// GEMM1: h1 = relu(dnn_in[B,512] @ W1[512,256] + b1) -> f16 [B,256]
// 32-row strip in LDS; 8 waves, each 2(m) x 2(n) register tile of 16x16 WMMAs.
// Output restaged through LDS for coalesced b128 stores.
// ---------------------------------------------------------------------------
__global__ void gemm1_relu(const _Float16* __restrict__ A,
                           const _Float16* __restrict__ Bp,
                           const float*    __restrict__ bias,
                           _Float16*       __restrict__ h1) {
    __shared__ __align__(16) _Float16 As[32 * KPAD];     // 32 KB
    const int m0 = blockIdx.x * 32;
    {
        const uint4* g = (const uint4*)(A + (size_t)m0 * KPAD);   // 4096 uint4
        uint4* sm = (uint4*)As;
#pragma unroll
        for (int t = 0; t < 16; ++t)
            sm[threadIdx.x + t * 256] = g[threadIdx.x + t * 256];
    }
    __syncthreads();

    const int wave = threadIdx.x >> 5;
    const int lane = threadIdx.x & 31;
    const int h    = lane >> 4;
    const int r    = lane & 15;
    const int ntA  = wave * 2, ntB = ntA + 1;

    v8f a00 = {}, a01 = {}, a10 = {}, a11 = {};
#pragma unroll 4
    for (int kb = 0; kb < KB1; ++kb) {
        const int k0 = kb * 32;
        v16h af0, af1;
#pragma unroll
        for (int v = 0; v < 8; ++v) {
            const int k = k0 + ((v >> 2) << 4) + (h << 3) + ((v & 3) << 1);
            af0[2 * v]     = As[r * KPAD + k];
            af0[2 * v + 1] = As[r * KPAD + k + 1];
            af1[2 * v]     = As[(16 + r) * KPAD + k];
            af1[2 * v + 1] = As[(16 + r) * KPAD + k + 1];
        }
        const v16h bf0 = *(const v16h*)(Bp + (((size_t)kb * NT1 + ntA) * 32 + lane) * 16);
        const v16h bf1 = *(const v16h*)(Bp + (((size_t)kb * NT1 + ntB) * 32 + lane) * 16);
        a00 = __builtin_amdgcn_wmma_f32_16x16x32_f16(false, af0, false, bf0, (short)0, a00, false, false);
        a01 = __builtin_amdgcn_wmma_f32_16x16x32_f16(false, af0, false, bf1, (short)0, a01, false, false);
        a10 = __builtin_amdgcn_wmma_f32_16x16x32_f16(false, af1, false, bf0, (short)0, a10, false, false);
        a11 = __builtin_amdgcn_wmma_f32_16x16x32_f16(false, af1, false, bf1, (short)0, a11, false, false);
    }
    __syncthreads();                                     // reuse As as out stage

    _Float16* Osh = As;                                  // 32 x 256 f16
    const int nA = ntA * 16 + r, nB = nA + 16;
    const float bA = bias[nA], bBv = bias[nB];
#pragma unroll
    for (int i = 0; i < 8; ++i) {
        const int mlo = i + 8 * h, mhi = 16 + mlo;       // C/D layout rows
        Osh[mlo * H1D + nA] = (_Float16)fmaxf(a00[i] + bA,  0.f);
        Osh[mlo * H1D + nB] = (_Float16)fmaxf(a01[i] + bBv, 0.f);
        Osh[mhi * H1D + nA] = (_Float16)fmaxf(a10[i] + bA,  0.f);
        Osh[mhi * H1D + nB] = (_Float16)fmaxf(a11[i] + bBv, 0.f);
    }
    __syncthreads();
    {
        const uint4* sm = (const uint4*)Osh;             // 1024 uint4
        uint4* g = (uint4*)(h1 + (size_t)m0 * H1D);
#pragma unroll
        for (int t = 0; t < 4; ++t)
            g[threadIdx.x + t * 256] = sm[threadIdx.x + t * 256];
    }
}

// ---------------------------------------------------------------------------
// GEMM2 fused: h2 = relu(h1[B,256] @ W2[256,128] + b2);
//              out[b] = h2 . final_w + final_b + fm_part[b]
// ---------------------------------------------------------------------------
__global__ void gemm2_final(const _Float16* __restrict__ A,
                            const _Float16* __restrict__ Bp,
                            const float*    __restrict__ bias,
                            const float*    __restrict__ fw,
                            const float*    __restrict__ fb,
                            const float*    __restrict__ fm_part,
                            float*          __restrict__ out) {
    __shared__ __align__(16) _Float16 As[16 * H1D];      // 8 KB
    __shared__ float rowsum[16];

    const int m0 = blockIdx.x * 16;
    if (threadIdx.x < 16) rowsum[threadIdx.x] = 0.f;
    {
        const uint4* g = (const uint4*)(A + (size_t)m0 * H1D);    // 1024 uint4
        uint4* sm = (uint4*)As;
#pragma unroll
        for (int t = 0; t < 4; ++t)
            sm[threadIdx.x + t * 256] = g[threadIdx.x + t * 256];
    }
    __syncthreads();

    const int wave = threadIdx.x >> 5;
    const int lane = threadIdx.x & 31;
    const int h    = lane >> 4;
    const int r    = lane & 15;

    v8f acc = {};
#pragma unroll
    for (int kb = 0; kb < KB2; ++kb) {
        const int k0 = kb * 32;
        v16h a;
#pragma unroll
        for (int v = 0; v < 8; ++v) {
            const int k = k0 + ((v >> 2) << 4) + (h << 3) + ((v & 3) << 1);
            a[2 * v]     = As[r * H1D + k];
            a[2 * v + 1] = As[r * H1D + k + 1];
        }
        const v16h bf = *(const v16h*)(Bp + (((size_t)kb * NT2 + wave) * 32 + lane) * 16);
        acc = __builtin_amdgcn_wmma_f32_16x16x32_f16(false, a, false, bf, (short)0, acc, false, false);
    }

    const int n = wave * 16 + r;
    const float bv = bias[n], fv = fw[n];
#pragma unroll
    for (int i = 0; i < 8; ++i) {
        float t = fmaxf(acc[i] + bv, 0.f) * fv;
        t += __shfl_xor(t, 1, 32);
        t += __shfl_xor(t, 2, 32);
        t += __shfl_xor(t, 4, 32);
        t += __shfl_xor(t, 8, 32);
        if (r == 0) atomicAdd(&rowsum[i + 8 * h], t);    // ds_add_f32
    }
    __syncthreads();
    if (threadIdx.x < 16)
        out[m0 + threadIdx.x] = rowsum[threadIdx.x] + fb[0] + fm_part[m0 + threadIdx.x];
}

// ---------------------------------------------------------------------------
extern "C" void kernel_launch(void* const* d_in, const int* in_sizes, int n_in,
                              void* d_out, int out_size, void* d_ws, size_t ws_size,
                              hipStream_t stream) {
    const float* rep   = (const float*)d_in[0];
    const float* dx    = (const float*)d_in[1];
    const int*   ids   = (const int*)  d_in[2];
    const float* fm1t  = (const float*)d_in[3];
    const float* fm2t  = (const float*)d_in[4];
    const float* fm1w  = (const float*)d_in[5];
    const float* fm1b  = (const float*)d_in[6];
    const float* w1    = (const float*)d_in[7];
    const float* b1    = (const float*)d_in[8];
    const float* w2    = (const float*)d_in[9];
    const float* b2    = (const float*)d_in[10];
    const float* fw    = (const float*)d_in[11];
    const float* fb    = (const float*)d_in[12];
    float* out = (float*)d_out;

    char* ws = (char*)d_ws;
    _Float16* dnn_in = (_Float16*)ws;  ws += (size_t)BB * KPAD * 2;   // 16 MB
    _Float16* h1     = (_Float16*)ws;  ws += (size_t)BB * H1D * 2;    //  8 MB
    float*    fmp    = (float*)ws;     ws += (size_t)BB * 4;          // 64 KB
    _Float16* w1p    = (_Float16*)ws;  ws += (size_t)KPAD * H1D * 2;  // 256 KB
    _Float16* w2p    = (_Float16*)ws;                                 // 64 KB

    pack_w1_frag<<<(KB1 * NT1 * 32 * 16) / 256, 256, 0, stream>>>(w1, w1p);
    pack_w2_frag<<<(KB2 * NT2 * 32 * 16) / 256, 256, 0, stream>>>(w2, w2p);
    gather_pack_fm<<<BB / 8, 256, 0, stream>>>(rep, dx, ids, fm1t, fm2t,
                                               fm1w, fm1b, dnn_in, fmp);
    gemm1_relu<<<BB / 32, 256, 0, stream>>>(dnn_in, w1p, b1, h1);
    gemm2_final<<<BB / 16, 256, 0, stream>>>(h1, w2p, b2, fw, fb, fmp, out);
}
